// LearnedRouter_10883447128554
// MI455X (gfx1250) — compile-verified, matrix operand order checked by reading, not run on hardware
//
#include <hip/hip_runtime.h>
#include <hip/hip_bf16.h>
#include <math.h>

#define HIDDEN 4096
#define NEXP 64

typedef __attribute__((ext_vector_type(2)))  float  v2f;
typedef __attribute__((ext_vector_type(4)))  float  v4f;
typedef __attribute__((ext_vector_type(8)))  float  v8f;
typedef __attribute__((ext_vector_type(8)))  __bf16 v8bf;
typedef __attribute__((ext_vector_type(16))) __bf16 v16bf;

// ---------------------------------------------------------------------------
// Prep: split W (f32) into bf16 hi/lo once:  W = hi + lo + O(2^-18)
// hi = rn_bf16(w), lo = rn_bf16(w - hi). 1 MB total in d_ws, L2-resident.
// ---------------------------------------------------------------------------
__global__ __launch_bounds__(256) void wsplit_kernel(
    const float* __restrict__ W, __bf16* __restrict__ whi,
    __bf16* __restrict__ wlo, int n)
{
    int i = blockIdx.x * 256 + threadIdx.x;
    if (i < n) {
        float f = W[i];
        __bf16 h = (__bf16)f;
        whi[i] = h;
        wlo[i] = (__bf16)(f - (float)h);
    }
}

// ---------------------------------------------------------------------------
// GEMM: logits[T,64] = x[T,H] @ W[64,H]^T via split-bf16 3-term product
//   logits = xhi*whi + xhi*wlo + xlo*whi   (f32 accumulate in WMMA)
// using V_WMMA_F32_16X16X32_BF16 (K=32, the fast matrix path on CDNA5).
//
// One wave owns a 16-token x 64-expert tile: 4 f32 accumulators (32 VGPRs).
// bf16 A (16x32): lane l -> M=l%16; elements 0-7 = K kg8..kg8+7,
//                 elements 8-15 = K 16+kg8..16+kg8+7, kg8=(l/16)*8.
// bf16 B (32x16): lane l -> N=l%16; same K mapping. Both are 16B runs.
// x is loaded once as f32 (4 x b128/lane/step) and split in registers.
// ---------------------------------------------------------------------------
__global__ __launch_bounds__(256) void router_gemm_kernel(
    const float* __restrict__ x, const __bf16* __restrict__ whi,
    const __bf16* __restrict__ wlo, float* __restrict__ logits, int T)
{
    const int lane   = threadIdx.x & 31;
    const int waveId = blockIdx.x * 8 + (threadIdx.x >> 5);
    const int tokenBase = waveId * 16;
    if (tokenBase >= T) return;              // wave-uniform: EXEC stays all-ones

    const int n   = lane & 15;               // M for A, N for B
    const int kg8 = (lane >> 4) << 3;        // 0 or 8: K sub-run per lane half

    int rowA = tokenBase + n;
    if (rowA >= T) rowA = T - 1;             // clamp keeps EXEC full

    const float* xp = x + (size_t)rowA * HIDDEN + kg8;

    const __bf16* bh0 = whi + (size_t)(n)      * HIDDEN + kg8;
    const __bf16* bh1 = whi + (size_t)(16 + n) * HIDDEN + kg8;
    const __bf16* bh2 = whi + (size_t)(32 + n) * HIDDEN + kg8;
    const __bf16* bh3 = whi + (size_t)(48 + n) * HIDDEN + kg8;
    const __bf16* bl0 = wlo + (size_t)(n)      * HIDDEN + kg8;
    const __bf16* bl1 = wlo + (size_t)(16 + n) * HIDDEN + kg8;
    const __bf16* bl2 = wlo + (size_t)(32 + n) * HIDDEN + kg8;
    const __bf16* bl3 = wlo + (size_t)(48 + n) * HIDDEN + kg8;

    v8f acc0 = {}, acc1 = {}, acc2 = {}, acc3 = {};

    #pragma unroll 2
    for (int k = 0; k < HIDDEN; k += 32) {
        // ---- load 16 f32 of x for this lane (two 8-elem runs, 4 x b128)
        v4f f0 = *(const v4f*)(xp + k);
        v4f f1 = *(const v4f*)(xp + k + 4);
        v4f f2 = *(const v4f*)(xp + k + 16);
        v4f f3 = *(const v4f*)(xp + k + 20);

        // ---- in-register split into bf16 hi/lo (exact residual capture)
        v16bf ahi, alo;
        float fa[16];
        #pragma unroll
        for (int i = 0; i < 4; ++i) {
            fa[i] = f0[i]; fa[4 + i] = f1[i]; fa[8 + i] = f2[i]; fa[12 + i] = f3[i];
        }
        #pragma unroll
        for (int i = 0; i < 16; ++i) {
            __bf16 h = (__bf16)fa[i];
            ahi[i] = h;
            alo[i] = (__bf16)(fa[i] - (float)h);
        }

        // ---- load bf16 W tiles: per tile two 16B runs -> v16bf
        #define LD16BF(p) __builtin_shufflevector(                         \
            *(const v8bf*)((p) + k), *(const v8bf*)((p) + k + 16),         \
            0, 1, 2, 3, 4, 5, 6, 7, 8, 9, 10, 11, 12, 13, 14, 15)
        v16bf Bh0 = LD16BF(bh0), Bh1 = LD16BF(bh1), Bh2 = LD16BF(bh2), Bh3 = LD16BF(bh3);
        v16bf Bl0 = LD16BF(bl0), Bl1 = LD16BF(bl1), Bl2 = LD16BF(bl2), Bl3 = LD16BF(bl3);
        #undef LD16BF

        // ---- 12 WMMAs, term-major so same-acc chains are distance 4
        acc0 = __builtin_amdgcn_wmma_f32_16x16x32_bf16(false, ahi, false, Bh0, (short)0, acc0, false, false);
        acc1 = __builtin_amdgcn_wmma_f32_16x16x32_bf16(false, ahi, false, Bh1, (short)0, acc1, false, false);
        acc2 = __builtin_amdgcn_wmma_f32_16x16x32_bf16(false, ahi, false, Bh2, (short)0, acc2, false, false);
        acc3 = __builtin_amdgcn_wmma_f32_16x16x32_bf16(false, ahi, false, Bh3, (short)0, acc3, false, false);
        acc0 = __builtin_amdgcn_wmma_f32_16x16x32_bf16(false, ahi, false, Bl0, (short)0, acc0, false, false);
        acc1 = __builtin_amdgcn_wmma_f32_16x16x32_bf16(false, ahi, false, Bl1, (short)0, acc1, false, false);
        acc2 = __builtin_amdgcn_wmma_f32_16x16x32_bf16(false, ahi, false, Bl2, (short)0, acc2, false, false);
        acc3 = __builtin_amdgcn_wmma_f32_16x16x32_bf16(false, ahi, false, Bl3, (short)0, acc3, false, false);
        acc0 = __builtin_amdgcn_wmma_f32_16x16x32_bf16(false, alo, false, Bh0, (short)0, acc0, false, false);
        acc1 = __builtin_amdgcn_wmma_f32_16x16x32_bf16(false, alo, false, Bh1, (short)0, acc1, false, false);
        acc2 = __builtin_amdgcn_wmma_f32_16x16x32_bf16(false, alo, false, Bh2, (short)0, acc2, false, false);
        acc3 = __builtin_amdgcn_wmma_f32_16x16x32_bf16(false, alo, false, Bh3, (short)0, acc3, false, false);
    }

    // C/D 16x16 f32 layout: lanes 0-15 -> M=v, N=lane; lanes 16-31 -> M=v+8.
    const int mOfs = (lane >> 4) ? 8 : 0;
    #pragma unroll
    for (int v = 0; v < 8; ++v) {
        int row = tokenBase + mOfs + v;
        if (row < T) {
            float* o = logits + (size_t)row * NEXP + n;
            o[0]  = acc0[v];
            o[16] = acc1[v];
            o[32] = acc2[v];
            o[48] = acc3[v];
        }
    }
}

// ---------------------------------------------------------------------------
// Softmax over E=64 + top-2 + L2-normalize. One wave32 per token; each lane
// owns 2 experts. Butterfly reductions via __shfl_xor (warpSize == 32).
// ---------------------------------------------------------------------------
__global__ __launch_bounds__(256) void router_softmax_topk_kernel(
    const float* __restrict__ logits, float* __restrict__ scores,
    float* __restrict__ ew, float* __restrict__ ei, int T)
{
    const int lane = threadIdx.x & 31;
    const int t = blockIdx.x * 8 + (threadIdx.x >> 5);
    if (t >= T) return;

    v2f l = *(const v2f*)(logits + (size_t)t * NEXP + 2 * lane);

    float m = fmaxf(l.x, l.y);
    #pragma unroll
    for (int off = 16; off; off >>= 1) m = fmaxf(m, __shfl_xor(m, off, 32));

    float e0 = __expf(l.x - m), e1 = __expf(l.y - m);
    float s = e0 + e1;
    #pragma unroll
    for (int off = 16; off; off >>= 1) s += __shfl_xor(s, off, 32);

    float inv = 1.0f / s;
    float s0 = e0 * inv, s1 = e1 * inv;
    v2f sc; sc.x = s0; sc.y = s1;
    *(v2f*)(scores + (size_t)t * NEXP + 2 * lane) = sc;

    float v1, w2; int i1, i2;
    if (s0 >= s1) { v1 = s0; i1 = 2 * lane;     w2 = s1; i2 = 2 * lane + 1; }
    else          { v1 = s1; i1 = 2 * lane + 1; w2 = s0; i2 = 2 * lane;     }

    #pragma unroll
    for (int off = 1; off < 32; off <<= 1) {
        float ov1 = __shfl_xor(v1, off, 32); int oi1 = __shfl_xor(i1, off, 32);
        float ov2 = __shfl_xor(w2, off, 32); int oi2 = __shfl_xor(i2, off, 32);
        float nv1, lo1; int ni1, li1;
        if (ov1 > v1) { nv1 = ov1; ni1 = oi1; lo1 = v1;  li1 = i1;  }
        else          { nv1 = v1;  ni1 = i1;  lo1 = ov1; li1 = oi1; }
        float hv2; int hi2;
        if (ov2 > w2) { hv2 = ov2; hi2 = oi2; } else { hv2 = w2; hi2 = i2; }
        if (hv2 > lo1) { w2 = hv2; i2 = hi2; } else { w2 = lo1; i2 = li1; }
        v1 = nv1; i1 = ni1;
    }

    if (lane == 0) {
        float nrm = rsqrtf(v1 * v1 + w2 * w2);
        ew[(size_t)t * 2]     = v1 * nrm;
        ew[(size_t)t * 2 + 1] = w2 * nrm;
        ei[(size_t)t * 2]     = (float)i1;
        ei[(size_t)t * 2 + 1] = (float)i2;
    }
}

extern "C" void kernel_launch(void* const* d_in, const int* in_sizes, int n_in,
                              void* d_out, int out_size, void* d_ws, size_t ws_size,
                              hipStream_t stream) {
    const float* x = (const float*)d_in[0];
    const float* W = (const float*)d_in[1];
    float* out = (float*)d_out;

    const int T = in_sizes[0] / HIDDEN;      // 16384 tokens

    float* scores = out;                              // [T, 64]
    float* logits = out + (size_t)T * NEXP;           // [T, 64]
    float* ew     = out + (size_t)2 * T * NEXP;       // [T, 2]
    float* ei     = ew + (size_t)T * 2;               // [T, 2]

    // W split scratch in d_ws (2 x 64*4096 bf16 = 1 MB)
    __bf16* whi = (__bf16*)d_ws;
    __bf16* wlo = whi + (size_t)NEXP * HIDDEN;

    const int nW = NEXP * HIDDEN;
    wsplit_kernel<<<(nW + 255) / 256, 256, 0, stream>>>(W, whi, wlo, nW);

    const int wavesGemm  = (T + 15) / 16;             // 16 tokens per wave
    const int blocksGemm = (wavesGemm + 7) / 8;       // 8 waves per block
    router_gemm_kernel<<<blocksGemm, 256, 0, stream>>>(x, whi, wlo, logits, T);

    const int blocksSm = (T + 7) / 8;                 // 1 wave per token
    router_softmax_topk_kernel<<<blocksSm, 256, 0, stream>>>(logits, scores, ew, ei, T);
}
